// PHMLinear_5720896438846
// MI455X (gfx1250) — matrix operand order repaired and checked by reading, not compiled, not run on hardware
//
#include <hip/hip_runtime.h>

// ---------------------------------------------------------------------------
// PHM linear + sigmoid gate, fused, for gfx1250 (MI455X).
//   W = sum_n kron(A[n], S[n])            [512 x 512]
//   out = x @ W^T + bias                  [65536 x 512]
//   res = out * sigmoid(out @ Gw^T + gb)
//
// Precision: GEMM1 uses bf16 hi/lo split (3x v_wmma_f32_16x16x32_bf16,
// fp32 accumulate) => ~fp32-class accuracy. GEMM2 (inside sigmoid) uses
// plain bf16 WMMA. Fusion keeps `out` in LDS so HBM traffic is just
// x-read + result-write (~256 MB => ~11 us floor at 23.3 TB/s).
// ---------------------------------------------------------------------------

typedef __attribute__((ext_vector_type(16))) __bf16 v16bf;
typedef __attribute__((ext_vector_type(8)))  __bf16 v8bf;
typedef __attribute__((ext_vector_type(8)))  float  v8f;

#define IN_F   512
#define OUT_F  512
#define TOKB   64          // tokens per workgroup
#define NTHR   256         // 8 waves (wave32)

static __device__ __forceinline__ unsigned short f2bf(float f) {
    unsigned u = __builtin_bit_cast(unsigned, f);
    unsigned r = u + 0x7FFFu + ((u >> 16) & 1u);   // round-to-nearest-even
    return (unsigned short)(r >> 16);
}
static __device__ __forceinline__ float bf2f(unsigned short h) {
    return __builtin_bit_cast(float, ((unsigned)h) << 16);
}

struct v8bf_pair { v8bf lo, hi; };
static __device__ __forceinline__ v16bf cat16(v8bf a, v8bf b) {
    v8bf_pair p{a, b};
    return __builtin_bit_cast(v16bf, p);
}

static __device__ __forceinline__ v8f wmma_bf16(v16bf a, v16bf b, v8f c) {
    // D = A(16x32) * B(32x16) + C, fp32 accumulate
    return __builtin_amdgcn_wmma_f32_16x16x32_bf16(
        false, a, false, b, (short)0, c, false, false);
}

// ---------------------------------------------------------------------------
// Prep: build W (row-major [out][in]) from A,S; emit bf16 hi/lo of W and
// bf16 of gate_w into workspace. 2048 blocks x 256 threads.
// ---------------------------------------------------------------------------
__global__ __launch_bounds__(NTHR)
void phm_prep(const float* __restrict__ A, const float* __restrict__ S,
              const float* __restrict__ Gw,
              unsigned short* __restrict__ Whi, unsigned short* __restrict__ Wlo,
              unsigned short* __restrict__ Ghi)
{
    int idx = blockIdx.x * NTHR + threadIdx.x;
    if (blockIdx.x < 1024) {                         // W elements: 512*512
        int o = idx >> 9, k = idx & 511;
        int i = o >> 7, p = o & 127;
        int j = k >> 7, q = k & 127;
        float w = 0.f;
#pragma unroll
        for (int n = 0; n < 4; ++n)
            w += A[n * 16 + i * 4 + j] * S[n * 16384 + p * 128 + q];
        unsigned short h = f2bf(w);
        Whi[idx] = h;
        Wlo[idx] = f2bf(w - bf2f(h));
    } else {                                         // gate_w -> bf16
        int id2 = idx - 262144;
        Ghi[id2] = f2bf(Gw[id2]);
    }
}

// ---------------------------------------------------------------------------
// Fused main kernel. Grid = 65536/64 = 1024 blocks, 256 threads, 256 KB LDS.
// LDS layout (bytes):
//   [0      , 65536 ) xhi  bf16 [64][512]   (reused as out_bf16 for GEMM2)
//   [65536  , 131072) xlo  bf16 [64][512]
//   [131072 , 262144) outf f32  [64][512]
// ---------------------------------------------------------------------------
__global__ __launch_bounds__(NTHR)
void phm_fused(const float* __restrict__ x,
               const unsigned short* __restrict__ Whi,
               const unsigned short* __restrict__ Wlo,
               const unsigned short* __restrict__ Ghi,
               const float* __restrict__ bias,
               const float* __restrict__ gate_b,
               float* __restrict__ out)
{
    extern __shared__ char smem[];
    unsigned short* xhi  = (unsigned short*)(smem);
    unsigned short* xlo  = (unsigned short*)(smem + 65536);
    float*          outf = (float*)(smem + 131072);
    unsigned short* obf  = (unsigned short*)(smem);      // overlays xhi

    const int tid    = threadIdx.x;
    const int lane   = tid & 31;
    const int wave   = tid >> 5;
    const int lhalf  = lane >> 4;       // 0: lanes 0-15, 1: lanes 16-31
    const int l15    = lane & 15;
    const size_t mbase = (size_t)blockIdx.x * TOKB;

    // ---- Phase 0: load x tile (64x512 f32), split into bf16 hi + lo ----
    {
        const float4* xg = (const float4*)(x + mbase * IN_F);
#pragma unroll 4
        for (int i = 0; i < 32; ++i) {
            int e4 = tid + i * NTHR;                 // 0..8191 float4s
            float4 v = xg[e4];
            ushort4 h, l;
            h.x = f2bf(v.x); l.x = f2bf(v.x - bf2f(h.x));
            h.y = f2bf(v.y); l.y = f2bf(v.y - bf2f(h.y));
            h.z = f2bf(v.z); l.z = f2bf(v.z - bf2f(h.z));
            h.w = f2bf(v.w); l.w = f2bf(v.w - bf2f(h.w));
            ((ushort4*)xhi)[e4] = h;
            ((ushort4*)xlo)[e4] = l;
        }
    }
    __syncthreads();

    // ---- Phase 1: out = x @ W^T + bias  (bf16x3 split precision) ----
    // Each wave owns 64 output columns: [wave*64, wave*64+64).
    for (int ns = 0; ns < 4; ++ns) {
        const int col = wave * 64 + ns * 16 + l15;   // B-fragment N = lane&15
        v8f acc[4] = {};
        const unsigned short* wh = Whi + (size_t)col * IN_F;
        const unsigned short* wl = Wlo + (size_t)col * IN_F;
        for (int ks = 0; ks < 16; ++ks) {
            const int k0 = ks * 32;
            // B fragment: per lane 16 consecutive K values (K0..15 / K16..31)
            v16bf bhi = *(const v16bf*)(wh + k0 + 16 * lhalf);
            v16bf blo = *(const v16bf*)(wl + k0 + 16 * lhalf);
#pragma unroll
            for (int m = 0; m < 4; ++m) {
                const int row = m * 16 + l15;        // A-fragment M = lane&15
                const unsigned short* ah = xhi + row * IN_F + k0 + 16 * lhalf;
                const unsigned short* al = xlo + row * IN_F + k0 + 16 * lhalf;
                v16bf Ahi = cat16(*(const v8bf*)(ah), *(const v8bf*)(ah + 8));
                v16bf Alo = cat16(*(const v8bf*)(al), *(const v8bf*)(al + 8));
                acc[m] = wmma_bf16(Ahi, bhi, acc[m]);   // hi*hi
                acc[m] = wmma_bf16(Ahi, blo, acc[m]);   // hi*lo
                acc[m] = wmma_bf16(Alo, bhi, acc[m]);   // lo*hi
            }
        }
        const float bv = bias[col];
#pragma unroll
        for (int m = 0; m < 4; ++m)
#pragma unroll
            for (int r = 0; r < 8; ++r) {
                int rowg = m * 16 + r + 8 * lhalf;   // C/D: lanes16-31 => M+8
                outf[rowg * OUT_F + col] = acc[m][r] + bv;
            }
    }
    __syncthreads();

    // ---- Phase 1.5: out (f32 LDS) -> bf16 A-operand buffer (reuses xhi) ----
    for (int j = tid; j < TOKB * OUT_F; j += NTHR)
        obf[j] = f2bf(outf[j]);
    __syncthreads();

    // ---- Phase 2: z = out @ Gw^T + gb ; res = out * sigmoid(z) ----
    for (int ns = 0; ns < 4; ++ns) {
        const int col = wave * 64 + ns * 16 + l15;
        v8f acc[4] = {};
        const unsigned short* gw = Ghi + (size_t)col * OUT_F;
        for (int ks = 0; ks < 16; ++ks) {
            const int k0 = ks * 32;
            v16bf b = *(const v16bf*)(gw + k0 + 16 * lhalf);
#pragma unroll
            for (int m = 0; m < 4; ++m) {
                const int row = m * 16 + l15;
                const unsigned short* ap = obf + row * OUT_F + k0 + 16 * lhalf;
                v16bf a = cat16(*(const v8bf*)(ap), *(const v8bf*)(ap + 8));
                acc[m] = wmma_bf16(a, b, acc[m]);
            }
        }
        const float gb = gate_b[col];
#pragma unroll
        for (int m = 0; m < 4; ++m)
#pragma unroll
            for (int r = 0; r < 8; ++r) {
                int rowg = m * 16 + r + 8 * lhalf;
                float z = acc[m][r] + gb;
                float g = 1.0f / (1.0f + __expf(-z));
                float o = outf[rowg * OUT_F + col];
                out[(mbase + rowg) * OUT_F + col] = o * g;
            }
    }
}

// ---------------------------------------------------------------------------
// Inputs (setup_inputs order): x, A, S, bias, gate_w, gate_b  (all fp32)
// Workspace: Whi (512KB) | Wlo (512KB) | Ghi (512KB)  = 1.5 MB of d_ws
// ---------------------------------------------------------------------------
extern "C" void kernel_launch(void* const* d_in, const int* in_sizes, int n_in,
                              void* d_out, int out_size, void* d_ws, size_t ws_size,
                              hipStream_t stream) {
    const float* x    = (const float*)d_in[0];
    const float* A    = (const float*)d_in[1];
    const float* S    = (const float*)d_in[2];
    const float* bias = (const float*)d_in[3];
    const float* Gw   = (const float*)d_in[4];
    const float* gb   = (const float*)d_in[5];
    float* out = (float*)d_out;

    unsigned short* Whi = (unsigned short*)d_ws;
    unsigned short* Wlo = Whi + 262144;
    unsigned short* Ghi = Wlo + 262144;

    phm_prep<<<2048, NTHR, 0, stream>>>(A, S, Gw, Whi, Wlo, Ghi);

    const int tokens = in_sizes[0] / IN_F;           // 65536
    const size_t lds_bytes = 262144;                 // 256 KB (<= 320 KB/WGP)
    phm_fused<<<tokens / TOKB, NTHR, lds_bytes, stream>>>(
        x, Whi, Wlo, Ghi, bias, gb, out);
}
